// ARTLearner_25838523253080
// MI455X (gfx1250) — matrix operand order repaired and checked by reading, not compiled, not run on hardware
//
#include <hip/hip_runtime.h>
#include <math.h>
#include <stdint.h>

// Problem sizes (match reference)
#define BB 128
#define TT 512
#define VV 32000
#define EE 512
#define HH 2048

typedef __bf16 bf16_t;
typedef __attribute__((ext_vector_type(8)))  __bf16 v8bf;
typedef __attribute__((ext_vector_type(16))) __bf16 v16bf;
typedef __attribute__((ext_vector_type(8)))  float  v8f;
typedef __attribute__((ext_vector_type(4)))  unsigned int u32x4;
typedef __attribute__((ext_vector_type(8)))  int i32x8;
typedef __attribute__((ext_vector_type(4)))  int i32x4;

// ---------- helpers ----------
__device__ __forceinline__ unsigned short f2bf(float f) {
  unsigned int u = __float_as_uint(f);
  u += 0x7FFFu + ((u >> 16) & 1u);   // round-to-nearest-even
  return (unsigned short)(u >> 16);
}

__device__ __forceinline__ v16bf cat16(v8bf lo, v8bf hi) {
  union { v16bf v; v8bf h[2]; } u;
  u.h[0] = lo; u.h[1] = hi;
  return u.v;
}

__device__ __forceinline__ v8bf ld8(const unsigned short* p) {
  return *(const v8bf*)p;   // 16B load; all call sites are 16B aligned
}

// ---------- Tensor Data Mover: stage a [n_rows=128] x [k=32 elems] bf16 tile
// from row-major W (row stride = k_len elems) into LDS at lds_byte.
// D# packing per CDNA5 ISA (2-D tile, groups 2/3 zero).
__device__ __forceinline__ void tdm_stage_B(unsigned lds_byte,
                                            const unsigned short* gsrc,
                                            unsigned k_len) {
  unsigned long long ga = (unsigned long long)(uintptr_t)gsrc;
  u32x4 g0 = {};
  g0[0] = 1u;                                   // count=1, is_restore=0, no gather
  g0[1] = lds_byte;                             // lds_addr [63:32]
  g0[2] = (unsigned)ga;                         // global_addr [95:64]
  g0[3] = (unsigned)((ga >> 32) & 0x01FFFFFFull) | 0x80000000u; // addr[120:96] | type=2
  i32x8 g1 = {};
  g1[0] = 0x00010000;                           // wg_mask=0, data_size=1 (2 bytes)
  g1[1] = (int)((k_len & 0xFFFFu) << 16);       // tensor_dim0 lo16 @ bits 48..63
  g1[2] = (int)(k_len >> 16);                   // tensor_dim0 hi16 | tensor_dim1 lo16 (=0)
  g1[3] = (int)(0x10u | (32u << 16));           // tensor_dim1 hi16 (dim1=1<<20) | tile_dim0=32
  g1[4] = 128;                                  // tile_dim1=128, tile_dim2=0
  g1[5] = (int)k_len;                           // tensor_dim0_stride lo32 (elements)
  g1[6] = 0;                                    // stride hi16 | dim1_stride lo16
  g1[7] = 0;
  i32x4 gz = {};
#if defined(__clang_major__) && (__clang_major__ >= 23)
  i32x8 gz8 = {};
  __builtin_amdgcn_tensor_load_to_lds(g0, g1, gz, gz, gz8, 0);
#else
  __builtin_amdgcn_tensor_load_to_lds(g0, g1, gz, gz, 0);
#endif
}

// ---------- pre-pass kernels ----------

// xs[t][b][0:E) = bf16(emb[ids[b][t]][0:E)) ; one thread = 8 contiguous elements
__global__ void gather_emb_kernel(const int* __restrict__ ids,
                                  const float* __restrict__ emb,
                                  unsigned short* __restrict__ xs) {
  long idx = (long)blockIdx.x * blockDim.x + threadIdx.x; // [0, T*B*(E/8))
  int e8 = (int)(idx & 63);
  int b  = (int)((idx >> 6) & 127);
  int t  = (int)(idx >> 13);
  int id = ids[(size_t)b * TT + t];
  const float* src = emb + (size_t)id * EE + (size_t)e8 * 8;
  unsigned short* dst = xs + ((size_t)t * BB + b) * EE + (size_t)e8 * 8;
#pragma unroll
  for (int i = 0; i < 8; ++i) dst[i] = f2bf(src[i]);
}

// D[row][col] = bf16( col<boundary ? S0[row][col] : S1[row][col-boundary] )
__global__ void convert_concat_kernel(const float* __restrict__ S0, int ld0,
                                      const float* __restrict__ S1, int ld1,
                                      int boundary, int ldd,
                                      unsigned short* __restrict__ D, long total) {
  long i = (long)blockIdx.x * blockDim.x + threadIdx.x;
  if (i >= total) return;
  long row = i / ldd;
  int  col = (int)(i - row * ldd);
  float v = (col < boundary) ? S0[row * (long)ld0 + col]
                             : S1[row * (long)ld1 + (col - boundary)];
  D[i] = f2bf(v);
}

__global__ void bias_sum_kernel(const float* __restrict__ a,
                                const float* __restrict__ b,
                                float* __restrict__ o, int n) {
  int i = blockIdx.x * blockDim.x + threadIdx.x;
  if (i < n) o[i] = a[i] + b[i];
}

__global__ void zero_u16_kernel(unsigned short* __restrict__ p, int n) {
  int i = blockIdx.x * blockDim.x + threadIdx.x;
  if (i < n) p[i] = 0;
}

// ---------- WMMA GEMM kernels ----------
// out[m][n] = act( sum_k A[m][k] * W[n][k] + bias[n] ), M fixed at 128.
// One block (8 waves) -> 128x128 tile; wave w -> rows [16w,16w+16).
// Shared B tile (32 K x 128 N) staged via TDM into double-buffered LDS.
// Per stage: all 16 ds_load_b128 issued first, then an 8-deep WMMA burst.

__device__ __forceinline__ v8bf load_dualA(const unsigned short* A0, int lda0,
                                           const unsigned short* A1, int lda1,
                                           int boundary, int row, int k) {
  const unsigned short* p = (k < boundary)
      ? A0 + (size_t)row * lda0 + k
      : A1 + (size_t)row * lda1 + (k - boundary);
  return ld8(p);
}

__global__ __launch_bounds__(256)
void wmma_gemm_bf16(const unsigned short* __restrict__ A0, int lda0,
                    const unsigned short* __restrict__ A1, int lda1,
                    int boundary, int K,
                    const unsigned short* __restrict__ W,   // [N][K] bf16
                    const float* __restrict__ bias,         // [N]
                    unsigned short* __restrict__ out, int ldout,
                    int act) {                              // 1=tanh, 2=relu
  __shared__ unsigned short smem[2][128][32];               // 2 x 8KB B tiles
  const int lane = threadIdx.x & 31;
  const int wave = threadIdx.x >> 5;
  const int half = lane >> 4;
  const int l16  = lane & 15;
  const int m0   = wave << 4;          // wave's 16 rows
  const int n0   = blockIdx.x << 7;    // block's 128 cols

  const unsigned lds_base = (unsigned)(uintptr_t)(void*)&smem[0][0][0];
  const unsigned short* Wblk = W + (size_t)n0 * K;

  v8f acc[8] = {};
  const int rowA = m0 + l16;
  const int nstages = K >> 5;

  if (wave == 0) tdm_stage_B(lds_base, Wblk, (unsigned)K);

  for (int s = 0; s < nstages; ++s) {
    const int p = s & 1;
    if (wave == 0) {
      if (s + 1 < nstages) {
        tdm_stage_B(lds_base + (unsigned)((p ^ 1) * 8192),
                    Wblk + ((s + 1) << 5), (unsigned)K);
        __builtin_amdgcn_s_wait_tensorcnt(1);   // stage s complete (in-order)
      } else {
        __builtin_amdgcn_s_wait_tensorcnt(0);
      }
    }
    __syncthreads();                            // B tile visible to all waves

    const int kk = s << 5;
    const int ka = kk + half * 8;
    v16bf afrag = cat16(load_dualA(A0, lda0, A1, lda1, boundary, rowA, ka),
                        load_dualA(A0, lda0, A1, lda1, boundary, rowA, ka + 16));
    const unsigned short* sb = &smem[p][0][0] + half * 8;
    v16bf bfr[8];
#pragma unroll
    for (int nt = 0; nt < 8; ++nt) {            // batch all LDS fragment loads
      const unsigned short* bp = sb + (((nt << 4) + l16) << 5);
      bfr[nt] = cat16(ld8(bp), ld8(bp + 16));
    }
#pragma unroll
    for (int nt = 0; nt < 8; ++nt)              // back-to-back WMMA burst
      acc[nt] = __builtin_amdgcn_wmma_f32_16x16x32_bf16(
          false, afrag, false, bfr[nt], (short)0, acc[nt], false, false);
    __syncthreads();                            // done reading buf p
  }

  const int rowC = m0 + half * 8;
#pragma unroll
  for (int nt = 0; nt < 8; ++nt) {
    const int n = n0 + (nt << 4) + l16;
    const float bn = bias[n];
#pragma unroll
    for (int v = 0; v < 8; ++v) {
      float val = acc[nt][v] + bn;
      val = (act == 1) ? tanhf(val) : fmaxf(val, 0.0f);
      out[(size_t)(rowC + v) * ldout + n] = f2bf(val);
    }
  }
}

__global__ __launch_bounds__(256)
void wmma_gemm_f32(const unsigned short* __restrict__ A, int K, // [128][K] bf16
                   const unsigned short* __restrict__ W,        // [N][K] bf16
                   const float* __restrict__ bias,
                   float* __restrict__ out, int ldout) {
  __shared__ unsigned short smem[2][128][32];
  const int lane = threadIdx.x & 31;
  const int wave = threadIdx.x >> 5;
  const int half = lane >> 4;
  const int l16  = lane & 15;
  const int m0   = wave << 4;
  const int n0   = blockIdx.x << 7;

  const unsigned lds_base = (unsigned)(uintptr_t)(void*)&smem[0][0][0];
  const unsigned short* Wblk = W + (size_t)n0 * K;

  v8f acc[8] = {};
  const int rowA = m0 + l16;
  const int nstages = K >> 5;

  if (wave == 0) tdm_stage_B(lds_base, Wblk, (unsigned)K);

  for (int s = 0; s < nstages; ++s) {
    const int p = s & 1;
    if (wave == 0) {
      if (s + 1 < nstages) {
        tdm_stage_B(lds_base + (unsigned)((p ^ 1) * 8192),
                    Wblk + ((s + 1) << 5), (unsigned)K);
        __builtin_amdgcn_s_wait_tensorcnt(1);
      } else {
        __builtin_amdgcn_s_wait_tensorcnt(0);
      }
    }
    __syncthreads();

    const int kk = s << 5;
    const int ka = kk + half * 8;
    const unsigned short* ap = A + (size_t)rowA * K + ka;
    v16bf afrag = cat16(ld8(ap), ld8(ap + 16));
    const unsigned short* sb = &smem[p][0][0] + half * 8;
    v16bf bfr[8];
#pragma unroll
    for (int nt = 0; nt < 8; ++nt) {
      const unsigned short* bp = sb + (((nt << 4) + l16) << 5);
      bfr[nt] = cat16(ld8(bp), ld8(bp + 16));
    }
#pragma unroll
    for (int nt = 0; nt < 8; ++nt)
      acc[nt] = __builtin_amdgcn_wmma_f32_16x16x32_bf16(
          false, afrag, false, bfr[nt], (short)0, acc[nt], false, false);
    __syncthreads();
  }

  const int rowC = m0 + half * 8;
#pragma unroll
  for (int nt = 0; nt < 8; ++nt) {
    const int n = n0 + (nt << 4) + l16;
    const float bn = bias[n];
#pragma unroll
    for (int v = 0; v < 8; ++v)
      out[(size_t)(rowC + v) * ldout + n] = acc[nt][v] + bn;
  }
}

// ---------- host launch ----------
extern "C" void kernel_launch(void* const* d_in, const int* in_sizes, int n_in,
                              void* d_out, int out_size, void* d_ws, size_t ws_size,
                              hipStream_t stream) {
  (void)in_sizes; (void)n_in; (void)out_size; (void)ws_size;

  const int*   ids  = (const int*)  d_in[0];
  const float* emb  = (const float*)d_in[1];
  const float* Wih0 = (const float*)d_in[2];
  const float* Whh0 = (const float*)d_in[3];
  const float* bih0 = (const float*)d_in[4];
  const float* bhh0 = (const float*)d_in[5];
  const float* Wih1 = (const float*)d_in[6];
  const float* Whh1 = (const float*)d_in[7];
  const float* bih1 = (const float*)d_in[8];
  const float* bhh1 = (const float*)d_in[9];
  const float* W1   = (const float*)d_in[10];
  const float* b1   = (const float*)d_in[11];
  const float* W2   = (const float*)d_in[12];
  const float* b2   = (const float*)d_in[13];
  float* out = (float*)d_out;

  // Workspace carve-up (~132 MB), 256B-aligned slices
  char* ws = (char*)d_ws;
  size_t off = 0;
  auto take = [&](size_t bytes) {
    char* p = ws + off;
    off = (off + bytes + 255) & ~(size_t)255;
    return p;
  };
  unsigned short* xs    = (unsigned short*)take((size_t)TT * BB * EE * 2);        // 64 MiB
  unsigned short* Wc0   = (unsigned short*)take((size_t)HH * (EE + HH) * 2);      // 10 MiB
  unsigned short* Wc1   = (unsigned short*)take((size_t)HH * (2 * HH) * 2);       // 16 MiB
  unsigned short* W1b   = (unsigned short*)take((size_t)EE * HH * 2);             //  2 MiB
  unsigned short* W2b   = (unsigned short*)take((size_t)VV * EE * 2);             // 31 MiB
  float*          bias0 = (float*)take((size_t)HH * 4);
  float*          bias1 = (float*)take((size_t)HH * 4);
  unsigned short* h0buf = (unsigned short*)take((size_t)2 * BB * HH * 2);         // double-buffered
  unsigned short* h1buf = (unsigned short*)take((size_t)2 * BB * HH * 2);
  unsigned short* zbuf  = (unsigned short*)take((size_t)BB * EE * 2);

  // --- pre-pass: gather embeddings (parallel over T), convert/concat weights, biases, h init ---
  gather_emb_kernel<<<(TT * BB * (EE / 8)) / 256, 256, 0, stream>>>(ids, emb, xs);

  {
    long tot0 = (long)HH * (EE + HH);
    convert_concat_kernel<<<(int)((tot0 + 255) / 256), 256, 0, stream>>>(
        Wih0, EE, Whh0, HH, EE, EE + HH, Wc0, tot0);
    long tot1 = (long)HH * (2 * HH);
    convert_concat_kernel<<<(int)((tot1 + 255) / 256), 256, 0, stream>>>(
        Wih1, HH, Whh1, HH, HH, 2 * HH, Wc1, tot1);
    long tW1 = (long)EE * HH;
    convert_concat_kernel<<<(int)((tW1 + 255) / 256), 256, 0, stream>>>(
        W1, HH, W1, HH, HH, HH, W1b, tW1);
    long tW2 = (long)VV * EE;
    convert_concat_kernel<<<(int)((tW2 + 255) / 256), 256, 0, stream>>>(
        W2, EE, W2, EE, EE, EE, W2b, tW2);
  }
  bias_sum_kernel<<<(HH + 255) / 256, 256, 0, stream>>>(bih0, bhh0, bias0, HH);
  bias_sum_kernel<<<(HH + 255) / 256, 256, 0, stream>>>(bih1, bhh1, bias1, HH);
  zero_u16_kernel<<<(BB * HH + 255) / 256, 256, 0, stream>>>(h0buf, BB * HH);
  zero_u16_kernel<<<(BB * HH + 255) / 256, 256, 0, stream>>>(h1buf, BB * HH);

  // --- serial scan: 2 WMMA GEMMs per step, weights resident in L2, B tiles via TDM ---
  for (int t = 0; t < TT; ++t) {
    const unsigned short* h0r = h0buf + (size_t)(t & 1) * BB * HH;
    unsigned short*       h0w = h0buf + (size_t)((t + 1) & 1) * BB * HH;
    const unsigned short* h1r = h1buf + (size_t)(t & 1) * BB * HH;
    unsigned short*       h1w = h1buf + (size_t)((t + 1) & 1) * BB * HH;
    const unsigned short* xt  = xs + (size_t)t * BB * EE;
    // h0 = tanh([x_t, h0] @ Wc0^T + bias0)
    wmma_gemm_bf16<<<HH / 128, 256, 0, stream>>>(
        xt, EE, h0r, HH, EE, EE + HH, Wc0, bias0, h0w, HH, 1);
    // h1 = tanh([h0, h1] @ Wc1^T + bias1)
    wmma_gemm_bf16<<<HH / 128, 256, 0, stream>>>(
        h0w, HH, h1r, HH, HH, 2 * HH, Wc1, bias1, h1w, HH, 1);
  }

  // --- head: z = relu(h1 @ W1^T + b1); out = z @ W2^T + b2 ---
  const unsigned short* h1fin = h1buf;  // after 512 steps final state is in buffer 0
  wmma_gemm_bf16<<<EE / 128, 256, 0, stream>>>(
      h1fin, HH, h1fin, HH, HH, HH, W1b, b1, zbuf, EE, 2);
  wmma_gemm_f32<<<VV / 128, 256, 0, stream>>>(zbuf, EE, W2b, b2, out, VV);
}
